// RSNNBlock_11751030522501
// MI455X (gfx1250) — matrix operand order repaired and checked by reading, not compiled, not run on hardware
//
#include <hip/hip_runtime.h>

// ---------------------------------------------------------------------------
// Fused GEMM (bf16x3 split on WMMA) + 16-step LIF epilogue for MI455X (gfx1250)
//   pre = x[8192,1024] @ W[2048,1024]^T + b
//   out = mean_t( heaviside(v_t - 1) ),  v <- 0.5*(v + pre), hard reset
//
// Pipeline: double-buffered f32 tiles staged with GLOBAL_LOAD_ASYNC_TO_LDS_B128
// (ASYNCcnt), hi/lo bf16 split done at fragment-load time (co-executes with the
// XDL/WMMA pipe), 24 v_wmma_f32_16x16x32_bf16 per K-step per wave.
// ---------------------------------------------------------------------------

typedef __attribute__((ext_vector_type(16))) __bf16 v16bf;
typedef __attribute__((ext_vector_type(8)))  float  v8f;
typedef __attribute__((ext_vector_type(4)))  float  f32x4;

namespace {
constexpr int Mdim = 8192;   // batch
constexpr int Ndim = 2048;   // hidden
constexpr int Kdim = 1024;   // in features
constexpr int BM = 128;
constexpr int BN = 128;
constexpr int BK = 32;       // one bf16 WMMA K-step
constexpr int LDF = 36;      // padded f32 row stride (144 B; gcd(36,64)=4 ->
                             // 16 distinct bank groups across 16 lanes)
constexpr int TILEF = BM * LDF;  // floats per (A or B) tile buffer
}

// Issue this thread's share (4+4 b128) of one 128x32 f32 A tile and B tile as
// async global->LDS copies. GVS addressing: saddr = tile row0, vaddr = byte off.
__device__ __forceinline__ void async_tile(unsigned long long gA, unsigned long long gB,
                                           unsigned ldsA, unsigned ldsB,
                                           int lr, int lc) {
    #pragma unroll
    for (int i = 0; i < 4; ++i) {
        const int r = lr + i * 32;
        const unsigned loff = (unsigned)(r * LDF + lc) * 4u;
        const unsigned goff = (unsigned)(r * Kdim + lc) * 4u;
        asm volatile("global_load_async_to_lds_b128 %0, %1, %2"
                     :: "v"(ldsA + loff), "v"(goff), "s"(gA) : "memory");
        asm volatile("global_load_async_to_lds_b128 %0, %1, %2"
                     :: "v"(ldsB + loff), "v"(goff), "s"(gB) : "memory");
    }
}

// Load 16 f32 from LDS (two 8-float runs) and split into bf16 hi + lo frags.
__device__ __forceinline__ void split16(const float* p0, const float* p1,
                                        v16bf& hi, v16bf& lo) {
    float v[16];
    *(f32x4*)(v + 0)  = *(const f32x4*)(p0);
    *(f32x4*)(v + 4)  = *(const f32x4*)(p0 + 4);
    *(f32x4*)(v + 8)  = *(const f32x4*)(p1);
    *(f32x4*)(v + 12) = *(const f32x4*)(p1 + 4);
    #pragma unroll
    for (int e = 0; e < 16; ++e) {
        const __bf16 h = (__bf16)v[e];
        hi[e] = h;
        lo[e] = (__bf16)(v[e] - (float)h);   // bf16->f32 is exact, residual is small
    }
}

__global__ __launch_bounds__(256)
void lif_gemm_kernel(const float* __restrict__ x, const float* __restrict__ W,
                     const float* __restrict__ b, float* __restrict__ out) {
    __shared__ float smem[4 * TILEF];   // [buf0: A,B][buf1: A,B]  = 72 KB

    const int tid  = threadIdx.x;
    const int lane = tid & 31;
    const int wave = tid >> 5;      // 0..7
    const int wm   = wave & 3;      // 4 wave rows  -> 32 M-rows each
    const int wn   = wave >> 2;     // 2 wave cols  -> 64 N-cols each
    const int bm0  = blockIdx.x * BM;
    const int bn0  = blockIdx.y * BN;

    // staging coordinates: 32 rows x (8 lanes * float4) per pass
    const int lr = tid >> 3;        // 0..31
    const int lc = (tid & 7) * 4;   // 0,4,...,28

    const unsigned lds0 = (unsigned)(size_t)smem;     // LDS byte offset of smem
    const unsigned long long gA0 = (unsigned long long)(x + (size_t)bm0 * Kdim);
    const unsigned long long gB0 = (unsigned long long)(W + (size_t)bn0 * Kdim);

    v8f acc[2][4] = {};             // 2 M-frags x 4 N-frags of 16x16 f32

    // prologue: stage K-tile 0 into buffer 0
    async_tile(gA0, gB0, lds0, lds0 + TILEF * 4u, lr, lc);

    for (int kk = 0; kk < Kdim; kk += BK) {
        const int cur = (kk / BK) & 1;
        const float* sA = smem + cur * 2 * TILEF;
        const float* sB = sA + TILEF;

        __syncthreads();            // prev compute done -> other buffer reusable
        if (kk + BK < Kdim) {
            const unsigned obase = lds0 + (unsigned)((1 - cur) * 2 * TILEF) * 4u;
            async_tile(gA0 + (unsigned long long)(kk + BK) * 4u,
                       gB0 + (unsigned long long)(kk + BK) * 4u,
                       obase, obase + TILEF * 4u, lr, lc);
            // 8 newer async ops outstanding -> current tile's 8 are complete
            asm volatile("s_wait_asynccnt 0x8" ::: "memory");
        } else {
            asm volatile("s_wait_asynccnt 0x0" ::: "memory");
        }
        __syncthreads();            // all waves' async data for current tile in LDS

        // ---- fragments + hi/lo split (CDNA5 16-bit WMMA layouts) ----
        v16bf Afh[2], Afl[2], Bfh[4], Bfl[4];
        // A 16x32: lane<16 holds K 0..7 & 16..23; lane>=16 holds 8..15 & 24..31
        const int kbA = (lane >> 4) * 8;
        #pragma unroll
        for (int mf = 0; mf < 2; ++mf) {
            const float* p = sA + (wm * 32 + mf * 16 + (lane & 15)) * LDF + kbA;
            split16(p, p + 16, Afh[mf], Afl[mf]);
        }
        // B 32x16: lane<16 holds K 0..7 & 16..23; lane>=16 holds 8..15 & 24..31
        const int kbB = (lane >> 4) * 8;
        #pragma unroll
        for (int nf = 0; nf < 4; ++nf) {
            const float* p = sB + (wn * 64 + nf * 16 + (lane & 15)) * LDF + kbB;
            split16(p, p + 16, Bfh[nf], Bfl[nf]);
        }

        // ---- bf16x3 split-precision MAC: hi*hi + hi*lo + lo*hi ----
        #pragma unroll
        for (int mf = 0; mf < 2; ++mf) {
            #pragma unroll
            for (int nf = 0; nf < 4; ++nf) {
                acc[mf][nf] = __builtin_amdgcn_wmma_f32_16x16x32_bf16(
                    false, Afh[mf], false, Bfh[nf], (short)0, acc[mf][nf], false, false);
                acc[mf][nf] = __builtin_amdgcn_wmma_f32_16x16x32_bf16(
                    false, Afh[mf], false, Bfl[nf], (short)0, acc[mf][nf], false, false);
                acc[mf][nf] = __builtin_amdgcn_wmma_f32_16x16x32_bf16(
                    false, Afl[mf], false, Bfh[nf], (short)0, acc[mf][nf], false, false);
            }
        }
    }

    // ---- epilogue: bias + 16 unrolled LIF steps entirely in registers ----
    // C/D layout: VGPR j holds M=j (lanes 0-15) / M=j+8 (lanes 16-31), N = lane%16
    const int ncolbase = bn0 + wn * 64 + (lane & 15);
    const int mrowbase = bm0 + wm * 32 + ((lane >> 4) << 3);
    #pragma unroll
    for (int mf = 0; mf < 2; ++mf) {
        #pragma unroll
        for (int nf = 0; nf < 4; ++nf) {
            const int ncol   = ncolbase + nf * 16;
            const float bias = b[ncol];
            #pragma unroll
            for (int j = 0; j < 8; ++j) {
                const float pre = acc[mf][nf][j] + bias;
                float v = 0.0f, s = 0.0f;
                #pragma unroll
                for (int t = 0; t < 16; ++t) {
                    v = 0.5f * (v + pre);                    // v += (pre - v)/tau
                    const float spk = (v >= 1.0f) ? 1.0f : 0.0f;
                    s += spk;
                    v -= v * spk;                            // hard reset to 0
                }
                out[(size_t)(mrowbase + mf * 16 + j) * Ndim + ncol] = s * 0.0625f;
            }
        }
    }
}

extern "C" void kernel_launch(void* const* d_in, const int* in_sizes, int n_in,
                              void* d_out, int out_size, void* d_ws, size_t ws_size,
                              hipStream_t stream) {
    const float* x = (const float*)d_in[0];   // [8192, 1024]
    const float* W = (const float*)d_in[1];   // [2048, 1024]
    const float* b = (const float*)d_in[2];   // [2048]
    float* out = (float*)d_out;               // [8192, 2048]
    (void)in_sizes; (void)n_in; (void)out_size; (void)d_ws; (void)ws_size;

    dim3 grid(Mdim / BM, Ndim / BN);          // 64 x 16 = 1024 workgroups
    lif_gemm_kernel<<<grid, 256, 0, stream>>>(x, W, b, out);
}
